// RNNModel_nodense_49048526520846
// MI455X (gfx1250) — matrix operand order repaired and checked by reading, not compiled, not run on hardware
//
#include <hip/hip_runtime.h>
#include <stddef.h>

// ---------------- types ----------------
typedef __attribute__((ext_vector_type(16))) __bf16 v16bf;
typedef __attribute__((ext_vector_type(8)))  __bf16 v8bf;
typedef __attribute__((ext_vector_type(4)))  __bf16 v4bf;
typedef __attribute__((ext_vector_type(8)))  float  v8f;

static __device__ __forceinline__ __bf16 f2bf(float f) {
    unsigned u = __builtin_bit_cast(unsigned, f);
    u += 0x7FFFu + ((u >> 16) & 1u);               // round-to-nearest-even
    unsigned short h = (unsigned short)(u >> 16);
    return __builtin_bit_cast(__bf16, h);
}

// LDS offset of a __shared__ object: flat-address low 32 bits are the LDS
// offset (the high 32 bits select the shared aperture).
static __device__ __forceinline__ unsigned lds_off(const void* p) {
    return (unsigned)(size_t)p;
}

// Async global->LDS copy of 16 bytes (tracked by ASYNCcnt).
static __device__ __forceinline__ void async_copy_b128(unsigned lds_dst,
                                                       const void* gsrc) {
    unsigned long long ga = (unsigned long long)gsrc;
    asm volatile("global_load_async_to_lds_b128 %0, %1, off"
                 :: "v"(lds_dst), "v"(ga) : "memory");
}
static __device__ __forceinline__ void async_wait0() {
    asm volatile("s_wait_asynccnt 0x0" ::: "memory");
}

// ---------------- problem constants ----------------
#define BB   512      // batch
#define TT   64       // timesteps
#define DD   1024     // WIN*CIN
#define HH   512      // hidden
#define H3   1536     // 3*H
#define N2   512      // WIN*COUT

// ---------------- prep kernels ----------------
__global__ __launch_bounds__(256) void cvt_bf16_kernel(
    const float* __restrict__ s, __bf16* __restrict__ d, int n) {
    for (int i = blockIdx.x * blockDim.x + threadIdx.x; i < n;
         i += gridDim.x * blockDim.x)
        d[i] = f2bf(s[i]);
}

__global__ __launch_bounds__(256) void make_w2sum_kernel(
    const float* __restrict__ W2, __bf16* __restrict__ d, int n) {
    // d[k*N2+j] = bf16(W2[k*N2+j] + W2[(k+H)*N2+j]),  k < H
    for (int i = blockIdx.x * blockDim.x + threadIdx.x; i < n;
         i += gridDim.x * blockDim.x)
        d[i] = f2bf(W2[i] + W2[i + HH * N2]);
}

__global__ __launch_bounds__(256) void zero_h_kernel(
    float* __restrict__ hf, __bf16* __restrict__ hb, int n) {
    for (int i = blockIdx.x * blockDim.x + threadIdx.x; i < n;
         i += gridDim.x * blockDim.x) {
        hf[i] = 0.0f;
        hb[i] = f2bf(0.0f);
    }
}

// ---------------- generic GEMM: C = A @ B + bias (bf16 WMMA, f32 acc) ---------
// A: [M x K] (f32 converted on the fly, or bf16 staged via async-to-LDS)
// B: [K x N] bf16 row-major; C: [M x N] f32 row-major. M%128==0,N%64==0,K%32==0
template <bool A_F32>
__global__ __launch_bounds__(256) void gemm_bias_kernel(
    const void* __restrict__ Aptr, const __bf16* __restrict__ Bg,
    const float* __restrict__ bias, float* __restrict__ C,
    int M, int N, int K) {
    constexpr int BM = 128, BN = 64, BK = 32;
    __shared__ __bf16 As[BM][BK + 8];   // row-major (m, k), padded
    __shared__ __bf16 Bs[BN][BK + 8];   // TRANSPOSED (n, k), padded

    const int tid  = threadIdx.x;
    const int wave = tid >> 5;          // 0..7 -> m-tile
    const int lane = tid & 31;
    const int bm0  = blockIdx.x * BM;
    const int bn0  = blockIdx.y * BN;

    // fill assignments
    const int a_row = tid >> 1;         // 0..127
    const int a_col = (tid & 1) * 16;   // 0 / 16
    const int b_k   = tid >> 3;         // 0..31
    const int b_n   = (tid & 7) * 8;    // 0..56

    v8f acc[4] = {};

    for (int k0 = 0; k0 < K; k0 += BK) {
        // ---- stage A tile ----
        if (A_F32) {
            const float* A = (const float*)Aptr;
            const float* src = A + (size_t)(bm0 + a_row) * K + k0 + a_col;
            float4 f0 = ((const float4*)src)[0];
            float4 f1 = ((const float4*)src)[1];
            float4 f2 = ((const float4*)src)[2];
            float4 f3 = ((const float4*)src)[3];
            __bf16* dst = &As[a_row][a_col];
            dst[0]  = f2bf(f0.x); dst[1]  = f2bf(f0.y);
            dst[2]  = f2bf(f0.z); dst[3]  = f2bf(f0.w);
            dst[4]  = f2bf(f1.x); dst[5]  = f2bf(f1.y);
            dst[6]  = f2bf(f1.z); dst[7]  = f2bf(f1.w);
            dst[8]  = f2bf(f2.x); dst[9]  = f2bf(f2.y);
            dst[10] = f2bf(f2.z); dst[11] = f2bf(f2.w);
            dst[12] = f2bf(f3.x); dst[13] = f2bf(f3.y);
            dst[14] = f2bf(f3.z); dst[15] = f2bf(f3.w);
        } else {
            // bf16 A: async copy straight into LDS (no VGPR round-trip)
            const __bf16* A = (const __bf16*)Aptr;
            const __bf16* src = A + (size_t)(bm0 + a_row) * K + k0 + a_col;
            const unsigned dst = lds_off(&As[a_row][a_col]);
            async_copy_b128(dst,      src);
            async_copy_b128(dst + 16, src + 8);
        }
        // ---- stage B tile transposed ----
        {
            const __bf16* src = Bg + (size_t)(k0 + b_k) * N + bn0 + b_n;
            v8bf v = *(const v8bf*)src;
#pragma unroll
            for (int j = 0; j < 8; ++j) Bs[b_n + j][b_k] = v[j];
        }
        // prefetch next B tile into cache (global_prefetch_b8)
        if (k0 + BK < K)
            __builtin_prefetch(Bg + (size_t)(k0 + BK + b_k) * N + bn0 + b_n, 0, 3);

        if (!A_F32) async_wait0();      // async LDS writes visible pre-barrier
        __syncthreads();

        // ---- A fragment (per wave): row strip wave*16 ----
        const int m  = wave * 16 + (lane & 15);
        const int ka = (lane >> 4) * 8;
        v8bf alo = *(const v8bf*)&As[m][ka];
        v8bf ahi = *(const v8bf*)&As[m][ka + 16];
        v16bf a;
#pragma unroll
        for (int i = 0; i < 8; ++i) { a[i] = alo[i]; a[i + 8] = ahi[i]; }

        const int kb = (lane >> 4) * 8;
#pragma unroll
        for (int nt = 0; nt < 4; ++nt) {
            const int nc = nt * 16 + (lane & 15);
            v8bf blo = *(const v8bf*)&Bs[nc][kb];
            v8bf bhi = *(const v8bf*)&Bs[nc][kb + 16];
            v16bf b;
#pragma unroll
            for (int i = 0; i < 8; ++i) { b[i] = blo[i]; b[i + 8] = bhi[i]; }
            acc[nt] = __builtin_amdgcn_wmma_f32_16x16x32_bf16(
                false, a, false, b, (short)0, acc[nt], false, false);
        }
        __syncthreads();
    }

    // ---- epilogue: C = acc + bias ----
#pragma unroll
    for (int nt = 0; nt < 4; ++nt) {
        const int col = bn0 + nt * 16 + (lane & 15);
        const float bv = bias[col];
#pragma unroll
        for (int v = 0; v < 8; ++v) {
            const int row = bm0 + wave * 16 + v + 8 * (lane >> 4);
            C[(size_t)row * N + col] = acc[nt][v] + bv;
        }
    }
}

// ---------------- GRU step: hg = h @ U + b_rec, then gating ----------------
// Block tile: 64 batch rows x 32 H-cols (=> 96 hg cols: z|r|h slices)
__global__ __launch_bounds__(256) void gru_step_kernel(
    const __bf16* __restrict__ h_bf, const float* __restrict__ h_f,
    const __bf16* __restrict__ Ub, const float* __restrict__ b_rec,
    const float* __restrict__ xg, int t,
    float* __restrict__ hn_f, __bf16* __restrict__ hn_bf) {
    __shared__ __bf16 As[64][40];     // h tile (m,k), padded
    __shared__ __bf16 Bs[96][40];     // U tile TRANSPOSED (strip-col, k)
    __shared__ float  Hg[64][100];    // hg strip, padded

    const int tid  = threadIdx.x;
    const int wave = tid >> 5;
    const int lane = tid & 31;
    const int bm0  = blockIdx.x * 64;   // batch base
    const int j0   = blockIdx.y * 32;   // H-column base

    const int m_tile  = wave & 3;       // 0..3
    const int n_group = wave >> 2;      // 0..1 -> strip cols [48*g, 48*g+48)

    const int a_row = tid >> 2;         // 0..63
    const int a_col = (tid & 3) * 8;    // 0,8,16,24
    const int b_k   = tid >> 3;         // 0..31
    const int b_n   = (tid & 7) * 4;    // 0..28

    v8f acc[3] = {};

    for (int k0 = 0; k0 < HH; k0 += 32) {
        // stage A: h_bf16 64x32 via async global->LDS (ASYNCcnt path)
        async_copy_b128(lds_off(&As[a_row][a_col]),
                        h_bf + (size_t)(bm0 + a_row) * HH + k0 + a_col);
        // stage B: three 32x32 gate slices of U, transposed into Bs
#pragma unroll
        for (int g = 0; g < 3; ++g) {
            const __bf16* src =
                Ub + (size_t)(k0 + b_k) * H3 + g * HH + j0 + b_n;
            v4bf v = *(const v4bf*)src;
#pragma unroll
            for (int j = 0; j < 4; ++j) Bs[g * 32 + b_n + j][b_k] = v[j];
        }
        async_wait0();
        __syncthreads();

        const int m  = m_tile * 16 + (lane & 15);
        const int ka = (lane >> 4) * 8;
        v8bf alo = *(const v8bf*)&As[m][ka];
        v8bf ahi = *(const v8bf*)&As[m][ka + 16];
        v16bf a;
#pragma unroll
        for (int i = 0; i < 8; ++i) { a[i] = alo[i]; a[i + 8] = ahi[i]; }

        const int kb = (lane >> 4) * 8;
#pragma unroll
        for (int jt = 0; jt < 3; ++jt) {
            const int nc = n_group * 48 + jt * 16 + (lane & 15);
            v8bf blo = *(const v8bf*)&Bs[nc][kb];
            v8bf bhi = *(const v8bf*)&Bs[nc][kb + 16];
            v16bf b;
#pragma unroll
            for (int i = 0; i < 8; ++i) { b[i] = blo[i]; b[i + 8] = bhi[i]; }
            acc[jt] = __builtin_amdgcn_wmma_f32_16x16x32_bf16(
                false, a, false, b, (short)0, acc[jt], false, false);
        }
        __syncthreads();
    }

    // write hg strip (+ recurrent bias) into LDS
#pragma unroll
    for (int jt = 0; jt < 3; ++jt) {
        const int scol = n_group * 48 + jt * 16 + (lane & 15);
        const int g = scol >> 5, cig = scol & 31;
        const float bb = b_rec[g * HH + j0 + cig];
#pragma unroll
        for (int v = 0; v < 8; ++v)
            Hg[m_tile * 16 + v + 8 * (lane >> 4)][scol] = acc[jt][v] + bb;
    }
    __syncthreads();

    // gating: z,r,relu-candidate, blend  (Keras GRU reset_after=True)
    for (int e = tid; e < 64 * 32; e += 256) {
        const int m = e >> 5, n = e & 31;
        const int b = bm0 + m, j = j0 + n;
        const size_t xb = ((size_t)b * TT + t) * H3;
        const float xz = xg[xb + j];
        const float xr = xg[xb + HH + j];
        const float xh = xg[xb + 2 * HH + j];
        const float hz = Hg[m][n];
        const float hr = Hg[m][32 + n];
        const float hh = Hg[m][64 + n];
        const float z = 1.0f / (1.0f + __expf(-(xz + hz)));
        const float r = 1.0f / (1.0f + __expf(-(xr + hr)));
        float hc = xh + r * hh;
        hc = hc > 0.0f ? hc : 0.0f;                    // relu activation
        const float ho = h_f[(size_t)b * HH + j];
        const float hn = z * ho + (1.0f - z) * hc;
        hn_f[(size_t)b * HH + j]  = hn;
        hn_bf[(size_t)b * HH + j] = f2bf(hn);
    }
}

// ---------------- workspace layout (bytes) ----------------
#define WB_OFF   ((size_t)0)                    // W_in bf16: 1024*1536*2 = 3,145,728
#define UB_OFF   ((size_t)3145728)              // U bf16:    512*1536*2 = 1,572,864
#define W2S_OFF  ((size_t)4718592)              // W2sum bf16: 512*512*2 =   524,288
#define HF0_OFF  ((size_t)5242880)              // h f32 buf0: 512*512*4 = 1,048,576
#define HF1_OFF  ((size_t)6291456)              // h f32 buf1
#define HB0_OFF  ((size_t)7340032)              // h bf16 buf0: 524,288
#define HB1_OFF  ((size_t)7864320)              // h bf16 buf1
#define XG_OFF   ((size_t)8388608)              // xg f32: 512*64*1536*4 = 201,326,592
// total = 209,715,200 bytes (200 MiB)

extern "C" void kernel_launch(void* const* d_in, const int* in_sizes, int n_in,
                              void* d_out, int out_size, void* d_ws, size_t ws_size,
                              hipStream_t stream) {
    const float* x     = (const float*)d_in[0];   // [B,T,WIN,CIN] = [32768,1024]
    const float* W_in  = (const float*)d_in[1];   // [1024,1536]
    const float* b_in  = (const float*)d_in[2];   // [1536]
    const float* U     = (const float*)d_in[3];   // [512,1536]
    const float* b_rec = (const float*)d_in[4];   // [1536]
    const float* W2    = (const float*)d_in[5];   // [1024,512]
    const float* b2    = (const float*)d_in[6];   // [512]
    float* out = (float*)d_out;                   // [512,512] flat

    char* ws = (char*)d_ws;
    __bf16* Wb  = (__bf16*)(ws + WB_OFF);
    __bf16* Ub  = (__bf16*)(ws + UB_OFF);
    __bf16* W2s = (__bf16*)(ws + W2S_OFF);
    float*  hf0 = (float*)(ws + HF0_OFF);
    float*  hf1 = (float*)(ws + HF1_OFF);
    __bf16* hb0 = (__bf16*)(ws + HB0_OFF);
    __bf16* hb1 = (__bf16*)(ws + HB1_OFF);
    float*  xg  = (float*)(ws + XG_OFF);

    // 1) convert weights, build W2sum, zero h0
    cvt_bf16_kernel<<<1024, 256, 0, stream>>>(W_in, Wb, DD * H3);
    cvt_bf16_kernel<<<1024, 256, 0, stream>>>(U, Ub, HH * H3);
    make_w2sum_kernel<<<512, 256, 0, stream>>>(W2, W2s, HH * N2);
    zero_h_kernel<<<512, 256, 0, stream>>>(hf0, hb0, BB * HH);

    // 2) input projection: xg = x @ W_in + b_in   (M=32768, N=1536, K=1024)
    gemm_bias_kernel<true><<<dim3((BB * TT) / 128, H3 / 64), 256, 0, stream>>>(
        (const void*)x, Wb, b_in, xg, BB * TT, H3, DD);

    // 3) recurrence: 64 sequential steps, double-buffered h
    for (int t = 0; t < TT; ++t) {
        const __bf16* hbc = (t & 1) ? hb1 : hb0;
        const float*  hfc = (t & 1) ? hf1 : hf0;
        __bf16* hbn = (t & 1) ? hb0 : hb1;
        float*  hfn = (t & 1) ? hf0 : hf1;
        gru_step_kernel<<<dim3(BB / 64, HH / 32), 256, 0, stream>>>(
            hbc, hfc, Ub, b_rec, xg, t, hfn, hbn);
    }
    // after 64 steps final h lands in buffer 0

    // 4) output head: out = hT @ W2sum + b2   (M=512, N=512, K=512)
    gemm_bias_kernel<false><<<dim3(BB / 128, N2 / 64), 256, 0, stream>>>(
        (const void*)hb0, W2s, b2, out, BB, N2, HH);
}
